// GPT2ParallelSelfAttention_84739704750800
// MI455X (gfx1250) — compile-verified, matrix operand order checked by reading, not run on hardware
//
#include <hip/hip_runtime.h>

// ---------------------------------------------------------------------------
// GPT2 self-attention for gfx1250 (MI455X): bf16 WMMA, fp32 accumulate.
//   K1: QKV projection GEMM  -> Q/K/V bf16 workspace, head-major [B,NH,S,HN]
//   K2: flash attention; K-tile staged via Tensor Data Mover (TDM) using the
//       6-arg __builtin_amdgcn_tensor_load_to_lds (amdgpu-toolchain form),
//       with s_wait_tensorcnt; manual cooperative copy as fallback.
//   K3: output projection GEMM -> fp32 d_out
// ---------------------------------------------------------------------------

typedef __attribute__((ext_vector_type(16))) __bf16 v16bf;
typedef __attribute__((ext_vector_type(8)))  __bf16 v8bf;
typedef __attribute__((ext_vector_type(8)))  float  v8f;
typedef __attribute__((ext_vector_type(4)))  unsigned int v4u;
typedef __attribute__((ext_vector_type(8)))  int v8i;
typedef __attribute__((ext_vector_type(4)))  int v4i;

#define HID   2048
#define NHEAD 16
#define HDIM  128
#define BATCH 4
#define SEQ   2048
#define N3H   6144

__device__ __forceinline__ v8f wmma_bf16(v16bf a, v16bf b, v8f c) {
  return __builtin_amdgcn_wmma_f32_16x16x32_bf16(false, a, false, b,
                                                 (short)0, c, false, false);
}

// Load a 16x32 A-fragment (or symmetric B-fragment) row for this lane.
// p points at element (row_for_lane, kbase_for_lane); layout per ISA 7.12.2:
// elements 0..7 = K kbase..kbase+7, elements 8..15 = K kbase+16..kbase+23.
__device__ __forceinline__ v16bf load_frag(const __bf16* p) {
  v8bf lo = *(const v8bf*)(p);
  v8bf hi = *(const v8bf*)(p + 16);
  v16bf r;
#pragma unroll
  for (int j = 0; j < 8; ++j) { r[j] = lo[j]; r[j + 8] = hi[j]; }
  return r;
}

// ---------------------------------------------------------------------------
// Kernel 1: mixed = X @ Wqkv + b ; scatter to Q/K/V bf16 [B,NH,S,HN]
// Block tile 128(M) x 128(N), K-step 64. 8 waves: 4 over M x 2 over N.
// 16 WMMAs per wave between barrier pairs.
// ---------------------------------------------------------------------------
__global__ __launch_bounds__(256) void qkv_proj_kernel(
    const float* __restrict__ X,     // [B*S, HID]
    const float* __restrict__ W,     // [HID, N3H]
    const float* __restrict__ bias,  // [N3H]
    __bf16* __restrict__ Qb, __bf16* __restrict__ Kb, __bf16* __restrict__ Vb) {
  __shared__ __align__(16) __bf16 As[128 * 64];  // [m][k]
  __shared__ __align__(16) __bf16 Bs[128 * 64];  // [n][k]

  const int n0   = blockIdx.x * 128;
  const int m0   = blockIdx.y * 128;
  const int tid  = threadIdx.x;
  const int lane = tid & 31;
  const int wv   = tid >> 5;
  const int waveM = wv & 3;   // 32 rows each
  const int waveN = wv >> 2;  // 64 cols each
  const int l16  = lane & 15;
  const int hf   = lane >> 4;

  const int mlA = tid >> 1, kcA = (tid & 1) * 32;   // A loader coords
  const int klB = tid >> 2, ncB = (tid & 3) * 32;   // B loader coords

  v8f acc[2][4] = {};

  for (int k0 = 0; k0 < HID; k0 += 64) {
    {  // A tile: 128x64 fp32 -> bf16
      const float* src = X + (size_t)(m0 + mlA) * HID + k0 + kcA;
      __bf16* dst = As + mlA * 64 + kcA;
#pragma unroll
      for (int j = 0; j < 32; ++j) dst[j] = (__bf16)src[j];
      if (k0 + 64 < HID) __builtin_prefetch(src + 64, 0, 0);
    }
    {  // B tile: 64x128 fp32 -> bf16, transposed into [n][k]
      const float* src = W + (size_t)(k0 + klB) * N3H + n0 + ncB;
#pragma unroll
      for (int j = 0; j < 32; ++j) Bs[(size_t)(ncB + j) * 64 + klB] = (__bf16)src[j];
      if (k0 + 64 < HID) __builtin_prefetch(src + (size_t)64 * N3H, 0, 0);
    }
    __syncthreads();

    v16bf af[2][2], bf[4][2];
#pragma unroll
    for (int kk = 0; kk < 2; ++kk) {
#pragma unroll
      for (int mi = 0; mi < 2; ++mi)
        af[mi][kk] = load_frag(As + (size_t)(waveM * 32 + mi * 16 + l16) * 64 +
                               kk * 32 + hf * 8);
#pragma unroll
      for (int ni = 0; ni < 4; ++ni)
        bf[ni][kk] = load_frag(Bs + (size_t)(waveN * 64 + ni * 16 + l16) * 64 +
                               kk * 32 + hf * 8);
    }
#pragma unroll
    for (int kk = 0; kk < 2; ++kk)
#pragma unroll
      for (int mi = 0; mi < 2; ++mi)
#pragma unroll
        for (int ni = 0; ni < 4; ++ni)
          acc[mi][ni] = wmma_bf16(af[mi][kk], bf[ni][kk], acc[mi][ni]);
    __syncthreads();
  }

  // Epilogue: bias, scatter into Q/K/V head-major bf16
#pragma unroll
  for (int mi = 0; mi < 2; ++mi) {
#pragma unroll
    for (int ni = 0; ni < 4; ++ni) {
      const int n = n0 + waveN * 64 + ni * 16 + l16;
      const float bv = bias[n];
      const int sel = n >> 11;
      const int rem = n & 2047;
      const int hd  = rem >> 7;
      const int dd  = rem & 127;
      __bf16* dst = (sel == 0) ? Qb : ((sel == 1) ? Kb : Vb);
#pragma unroll
      for (int r = 0; r < 8; ++r) {
        const int m = m0 + waveM * 32 + mi * 16 + r + hf * 8;
        const int bb = m >> 11, ss = m & 2047;
        const float v = acc[mi][ni][r] + bv;
        dst[(((size_t)bb * NHEAD + hd) * SEQ + ss) * HDIM + dd] = (__bf16)v;
      }
    }
  }
}

// ---------------------------------------------------------------------------
// Kernel 2: flash attention. Grid (S/128, NH, B). 8 waves; each wave owns 16
// query rows. K blocks of 64 keys streamed through LDS with online softmax.
// K tile staged by the Tensor Data Mover when available.
// ---------------------------------------------------------------------------
__global__ __launch_bounds__(256) void attn_kernel(
    const __bf16* __restrict__ Qb, const __bf16* __restrict__ Kb,
    const __bf16* __restrict__ Vb, __bf16* __restrict__ Ctx) {
  __shared__ __align__(16) __bf16 Ks[64 * 128];   // [key][d]
  __shared__ __align__(16) __bf16 Vs[128 * 64];   // [d][key] (transposed)
  __shared__ __align__(16) __bf16 Ps[128 * 64];   // [q][key] wave-private slices

  const int qt = blockIdx.x, hd = blockIdx.y, bb = blockIdx.z;
  const int tid  = threadIdx.x;
  const int lane = tid & 31;
  const int wv   = tid >> 5;
  const int l16  = lane & 15;
  const int hf   = lane >> 4;
  const int qrow0 = qt * 128 + wv * 16;

  const size_t bh = (size_t)bb * NHEAD + hd;
  const __bf16* Qbase = Qb + bh * SEQ * HDIM;
  const __bf16* Kbase = Kb + bh * SEQ * HDIM;
  const __bf16* Vbase = Vb + bh * SEQ * HDIM;

  // Q fragments: 16 rows x 128 d = 4 k-steps, resident in registers.
  v16bf qf[4];
  {
    const __bf16* qrow = Qbase + (size_t)(qrow0 + l16) * HDIM + hf * 8;
#pragma unroll
    for (int kk = 0; kk < 4; ++kk) qf[kk] = load_frag(qrow + kk * 32);
  }

  float mrow[8], lrow[8];
#pragma unroll
  for (int r = 0; r < 8; ++r) { mrow[r] = -1e30f; lrow[r] = 0.0f; }
  v8f o[8] = {};

  const float scale = 0.08838834764831845f;  // 1/sqrt(128)
  const int nblk = 2 * qt + 2;               // causal: keys < (qt+1)*128

#if __has_builtin(__builtin_amdgcn_tensor_load_to_lds)
  // TDM descriptor group 1 (constant across K blocks). ISA 8.4 bit layout:
  // d0: data_size=1 (2B) at bits[17:16]
  // d1[31:16] = tensor_dim0 lo16 (=HDIM)
  // d2[31:16] = tensor_dim1 lo16 (=SEQ)
  // d3[31:16] = tile_dim0 (=HDIM)
  // d4[15:0]  = tile_dim1 (=64 keys)
  // d5        = tensor_dim0_stride lo32 (=HDIM)
  v8i tdm_g1;
  tdm_g1[0] = (int)(1u << 16);
  tdm_g1[1] = (int)((unsigned)HDIM << 16);
  tdm_g1[2] = (int)((unsigned)SEQ << 16);
  tdm_g1[3] = (int)((unsigned)HDIM << 16);
  tdm_g1[4] = 64;
  tdm_g1[5] = HDIM;
  tdm_g1[6] = 0;
  tdm_g1[7] = 0;
  const v4i tdm_z4 = {0, 0, 0, 0};
  const v8i tdm_z8 = {0, 0, 0, 0, 0, 0, 0, 0};
  const unsigned lds_ks = (unsigned)(uintptr_t)(&Ks[0]);  // LDS byte offset
#endif

  for (int blk = 0; blk < nblk; ++blk) {
    const int key0 = blk * 64;
    __syncthreads();  // previous block's LDS reads complete

#if __has_builtin(__builtin_amdgcn_tensor_load_to_lds)
    if (wv == 0) {
      // Group 0: count=1, lds_addr, 57-bit global tile address, type=2.
      const unsigned long long ga =
          (unsigned long long)(uintptr_t)(Kbase + (size_t)key0 * HDIM);
      v4u g0;
      g0[0] = 1u;                                            // count=1
      g0[1] = lds_ks;                                        // lds_addr
      g0[2] = (unsigned)(ga & 0xFFFFFFFFu);                  // global_addr lo
      g0[3] = (unsigned)((ga >> 32) & 0x01FFFFFFu) | (2u << 30);  // hi | type=2
      // amdgpu-toolchain (clang-23) 6-arg form:
      // (g0 v4u, g1 v8i, g2 v4i, g3 v4i, v8i, cpol)
      __builtin_amdgcn_tensor_load_to_lds(g0, tdm_g1, tdm_z4, tdm_z4, tdm_z8, 0);
    }
#else
    {  // fallback: cooperative K copy (b128 vector moves)
      const int ky = tid >> 2, dc = (tid & 3) * 32;
      const __bf16* ksrc = Kbase + (size_t)(key0 + ky) * HDIM + dc;
#pragma unroll
      for (int j = 0; j < 4; ++j)
        *(v8bf*)(Ks + (size_t)ky * 128 + dc + j * 8) = *(const v8bf*)(ksrc + j * 8);
    }
#endif

    {  // V tile: global -> registers -> transposed LDS [d][key]
      const int ky = tid >> 2, dc = (tid & 3) * 32;
      const __bf16* vsrc = Vbase + (size_t)(key0 + ky) * HDIM + dc;
#pragma unroll
      for (int j = 0; j < 32; ++j) Vs[(size_t)(dc + j) * 64 + ky] = vsrc[j];
    }

#if __has_builtin(__builtin_amdgcn_tensor_load_to_lds)
    if (wv == 0) {
#if __has_builtin(__builtin_amdgcn_s_wait_tensorcnt)
      __builtin_amdgcn_s_wait_tensorcnt(0);
#else
      asm volatile("s_wait_tensorcnt 0x0" ::: "memory");
#endif
    }
#endif
    __syncthreads();

    // Scores: 16 q rows x 64 keys
    v8f sc[4] = {};
#pragma unroll
    for (int t = 0; t < 4; ++t)
#pragma unroll
      for (int kk = 0; kk < 4; ++kk) {
        v16bf kbf = load_frag(Ks + (size_t)(t * 16 + l16) * 128 + kk * 32 + hf * 8);
        sc[t] = wmma_bf16(qf[kk], kbf, sc[t]);
      }

    // scale + causal mask (reference semantics: masked score == -10000)
    const bool domask = (blk >= 2 * qt);
#pragma unroll
    for (int t = 0; t < 4; ++t)
#pragma unroll
      for (int r = 0; r < 8; ++r) {
        float v = sc[t][r] * scale;
        if (domask) {
          const int qg = qrow0 + r + hf * 8;
          const int kg = key0 + t * 16 + l16;
          if (kg > qg) v = -10000.0f;
        }
        sc[t][r] = v;
      }

    // online softmax: row max / exp / row sum (rows live in 16-lane halves)
    float alpha[8];
#pragma unroll
    for (int r = 0; r < 8; ++r) {
      float mx = sc[0][r];
#pragma unroll
      for (int t = 1; t < 4; ++t) mx = fmaxf(mx, sc[t][r]);
#pragma unroll
      for (int off = 1; off < 16; off <<= 1)
        mx = fmaxf(mx, __shfl_xor(mx, off, 32));
      const float mnew = fmaxf(mrow[r], mx);
      alpha[r] = __expf(mrow[r] - mnew);
      mrow[r] = mnew;
      float rs = 0.0f;
#pragma unroll
      for (int t = 0; t < 4; ++t) {
        const float e = __expf(sc[t][r] - mnew);
        sc[t][r] = e;
        rs += e;
      }
#pragma unroll
      for (int off = 1; off < 16; off <<= 1)
        rs += __shfl_xor(rs, off, 32);
      lrow[r] = lrow[r] * alpha[r] + rs;
    }
#pragma unroll
    for (int di = 0; di < 8; ++di)
#pragma unroll
      for (int r = 0; r < 8; ++r) o[di][r] *= alpha[r];

    // C-layout -> A-layout via wave-private LDS slice
#pragma unroll
    for (int t = 0; t < 4; ++t)
#pragma unroll
      for (int r = 0; r < 8; ++r)
        Ps[(size_t)(wv * 16 + r + hf * 8) * 64 + t * 16 + l16] = (__bf16)sc[t][r];

    v16bf pf[2];
    {
      const __bf16* prow = Ps + (size_t)(wv * 16 + l16) * 64 + hf * 8;
      pf[0] = load_frag(prow);
      pf[1] = load_frag(prow + 32);
    }
#pragma unroll
    for (int di = 0; di < 8; ++di)
#pragma unroll
      for (int ks = 0; ks < 2; ++ks) {
        v16bf vf = load_frag(Vs + (size_t)(di * 16 + l16) * 64 + ks * 32 + hf * 8);
        o[di] = wmma_bf16(pf[ks], vf, o[di]);
      }
  }

  // epilogue: O / l -> ctx bf16 [B,S,H] (heads re-interleaved)
#pragma unroll
  for (int r = 0; r < 8; ++r) {
    const float inv = 1.0f / lrow[r];
    const int qg = qrow0 + r + hf * 8;
#pragma unroll
    for (int di = 0; di < 8; ++di) {
      const int dd = di * 16 + l16;
      Ctx[((size_t)bb * SEQ + qg) * HID + hd * HDIM + dd] = (__bf16)(o[di][r] * inv);
    }
  }
}

// ---------------------------------------------------------------------------
// Kernel 3: out = ctx(bf16) @ Wdense + b ; fp32 output
// ---------------------------------------------------------------------------
__global__ __launch_bounds__(256) void out_proj_kernel(
    const __bf16* __restrict__ Ctx,  // [B*S, HID] bf16
    const float* __restrict__ W,     // [HID, HID]
    const float* __restrict__ bias,  // [HID]
    float* __restrict__ Out) {       // [B*S, HID]
  __shared__ __align__(16) __bf16 As[128 * 64];  // [m][k]
  __shared__ __align__(16) __bf16 Bs[128 * 64];  // [n][k]

  const int n0   = blockIdx.x * 128;
  const int m0   = blockIdx.y * 128;
  const int tid  = threadIdx.x;
  const int lane = tid & 31;
  const int wv   = tid >> 5;
  const int waveM = wv & 3;
  const int waveN = wv >> 2;
  const int l16  = lane & 15;
  const int hf   = lane >> 4;

  const int mlA = tid >> 1, kcA = (tid & 1) * 32;
  const int klB = tid >> 2, ncB = (tid & 3) * 32;

  v8f acc[2][4] = {};

  for (int k0 = 0; k0 < HID; k0 += 64) {
    {  // A tile (already bf16): 4x 16B vector copies
      const __bf16* src = Ctx + (size_t)(m0 + mlA) * HID + k0 + kcA;
      __bf16* dst = As + (size_t)mlA * 64 + kcA;
#pragma unroll
      for (int j = 0; j < 4; ++j)
        *(v8bf*)(dst + j * 8) = *(const v8bf*)(src + j * 8);
      if (k0 + 64 < HID) __builtin_prefetch(src + 64, 0, 0);
    }
    {  // B tile: 64x128 fp32 -> bf16, transposed into [n][k]
      const float* src = W + (size_t)(k0 + klB) * HID + n0 + ncB;
#pragma unroll
      for (int j = 0; j < 32; ++j) Bs[(size_t)(ncB + j) * 64 + klB] = (__bf16)src[j];
      if (k0 + 64 < HID) __builtin_prefetch(src + (size_t)64 * HID, 0, 0);
    }
    __syncthreads();

    v16bf af[2][2], bf[4][2];
#pragma unroll
    for (int kk = 0; kk < 2; ++kk) {
#pragma unroll
      for (int mi = 0; mi < 2; ++mi)
        af[mi][kk] = load_frag(As + (size_t)(waveM * 32 + mi * 16 + l16) * 64 +
                               kk * 32 + hf * 8);
#pragma unroll
      for (int ni = 0; ni < 4; ++ni)
        bf[ni][kk] = load_frag(Bs + (size_t)(waveN * 64 + ni * 16 + l16) * 64 +
                               kk * 32 + hf * 8);
    }
#pragma unroll
    for (int kk = 0; kk < 2; ++kk)
#pragma unroll
      for (int mi = 0; mi < 2; ++mi)
#pragma unroll
        for (int ni = 0; ni < 4; ++ni)
          acc[mi][ni] = wmma_bf16(af[mi][kk], bf[ni][kk], acc[mi][ni]);
    __syncthreads();
  }

#pragma unroll
  for (int mi = 0; mi < 2; ++mi)
#pragma unroll
    for (int ni = 0; ni < 4; ++ni) {
      const int n = n0 + waveN * 64 + ni * 16 + l16;
      const float bv = bias[n];
#pragma unroll
      for (int r = 0; r < 8; ++r) {
        const int m = m0 + waveM * 32 + mi * 16 + r + hf * 8;
        Out[(size_t)m * HID + n] = acc[mi][ni][r] + bv;
      }
    }
}

// ---------------------------------------------------------------------------
// Launch
// ---------------------------------------------------------------------------
extern "C" void kernel_launch(void* const* d_in, const int* in_sizes, int n_in,
                              void* d_out, int out_size, void* d_ws, size_t ws_size,
                              hipStream_t stream) {
  (void)in_sizes; (void)n_in; (void)out_size; (void)ws_size;
  const float* hidden  = (const float*)d_in[0];
  // d_in[1] = ltor_mask: causal mask computed analytically, not read.
  const float* w_qkv   = (const float*)d_in[2];
  const float* b_qkv   = (const float*)d_in[3];
  const float* w_dense = (const float*)d_in[4];
  const float* b_dense = (const float*)d_in[5];

  char* ws = (char*)d_ws;
  const size_t QKV_BYTES = (size_t)BATCH * NHEAD * SEQ * HDIM * sizeof(__bf16);  // 32 MB
  __bf16* Qb  = (__bf16*)(ws);
  __bf16* Kb  = (__bf16*)(ws + QKV_BYTES);
  __bf16* Vb  = (__bf16*)(ws + 2 * QKV_BYTES);
  __bf16* Ctx = (__bf16*)(ws + 3 * QKV_BYTES);

  qkv_proj_kernel<<<dim3(N3H / 128, (BATCH * SEQ) / 128), 256, 0, stream>>>(
      hidden, w_qkv, b_qkv, Qb, Kb, Vb);
  attn_kernel<<<dim3(SEQ / 128, NHEAD, BATCH), 256, 0, stream>>>(Qb, Kb, Vb, Ctx);
  out_proj_kernel<<<dim3(HID / 128, (BATCH * SEQ) / 128), 256, 0, stream>>>(
      Ctx, w_dense, b_dense, (float*)d_out);
}
